// MultiAttention_24343874633815
// MI455X (gfx1250) — matrix operand order repaired and checked by reading, not compiled, not run on hardware
//
#include <hip/hip_runtime.h>
#include <hip/hip_bf16.h>
#include <stdint.h>

// MI455X / gfx1250 fused multi-head attention block.
//   x:[32768,8,256]f32 -> per-head QKV (bf16 WMMA, f32 acc) -> causal softmax
//   -> wei@V -> concat heads -> output projection -> [32768,8,256]f32
// All GEMMs on V_WMMA_F32_16X16X32_BF16 (wave32). Weights transposed to
// [n][k] bf16 in d_ws by a pre-kernel so B-fragments are contiguous loads.
// x tile staging uses GLOBAL_LOAD_ASYNC_TO_LDS_B128 (ASYNCcnt) into an LDS
// fp32 buffer that aliases the k/q regions (dead until after conversion).

#define DMODEL 256
#define DH     64
#define NHEADS 4
#define BATCH  32768
#define WAVES  4          // waves per block, 16 token-rows (2 words) per wave

#define USE_ASYNC_STAGE 1

typedef __attribute__((ext_vector_type(16))) __bf16 v16bf;
typedef __attribute__((ext_vector_type(8)))  float  v8f;

struct alignas(16) U4 { uint32_t x, y, z, w; };
struct alignas(16) F4 { float x, y, z, w; };

struct Frag {
  union { v16bf v; U4 q[2]; uint32_t w[8]; };
};

__device__ __forceinline__ uint16_t f2bf(float f) {
  uint32_t u = __float_as_uint(f);
  u = (u + 0x7fffu + ((u >> 16) & 1u)) >> 16;   // round-to-nearest-even
  return (uint16_t)u;
}
__device__ __forceinline__ uint32_t pk_bf16(float a, float b) {
  return (uint32_t)f2bf(a) | ((uint32_t)f2bf(b) << 16);
}
__device__ __forceinline__ v8f vzero() {
  v8f z = {0.f, 0.f, 0.f, 0.f, 0.f, 0.f, 0.f, 0.f};
  return z;
}

// A-fragment (16x32 bf16, M x K), base is row-major [m][256].
// Lane L: m = L&15, K = kb..kb+7 (VGPR0-3) and kb+16..kb+23 (VGPR4-7),
// kb = k0 + (L>>4)*8.  Two contiguous 16B loads.
__device__ __forceinline__ Frag load_a16(const uint16_t* base, int lane, int k0) {
  Frag f;
  const int m  = lane & 15;
  const int kb = k0 + ((lane >> 4) << 3);
  const uint16_t* p = base + m * 256 + kb;
  f.q[0] = *reinterpret_cast<const U4*>(p);
  f.q[1] = *reinterpret_cast<const U4*>(p + 16);
  return f;
}

// B-fragment (32x16 bf16, K x N), base stored row-major [n][256] (i.e. B^T).
// Lane L: n = n0 + (L&15), K = kb..kb+7 and kb+16..kb+23, kb = k0 + (L>>4)*8.
__device__ __forceinline__ Frag load_b16(const uint16_t* base, int lane, int n0, int k0) {
  Frag f;
  const int n  = n0 + (lane & 15);
  const int kb = k0 + ((lane >> 4) << 3);
  const uint16_t* p = base + (size_t)n * 256 + kb;
  f.q[0] = *reinterpret_cast<const U4*>(p);
  f.q[1] = *reinterpret_cast<const U4*>(p + 16);
  return f;
}

// B-fragment for wei@V: K = s (only 16 valid, upper 16 zero). vT is [64][16].
__device__ __forceinline__ Frag load_bv(const uint16_t* vT, int lane, int e0) {
  Frag f;
  {
    const uint16_t* p = vT + (e0 + (lane & 15)) * 16 + ((lane >> 4) << 3);
    f.q[0] = *reinterpret_cast<const U4*>(p);
    f.w[4] = f.w[5] = f.w[6] = f.w[7] = 0u;
  }
  return f;
}

// Scatter a f32 D-tile (lane col n, rows m=(L>>4)*8+r) into row-major bf16 LDS.
__device__ __forceinline__ void store_tile_rm(uint16_t* base, int lane, int n0,
                                              const v8f& acc) {
  const int col = n0 + (lane & 15);
  const int r0  = (lane >> 4) << 3;
  #pragma unroll
  for (int r = 0; r < 8; ++r) base[(r0 + r) * 256 + col] = f2bf(acc[r]);
}

// Store a V D-tile transposed: vT[e][s], contiguous 16B per lane.
__device__ __forceinline__ void store_tile_vT(uint16_t* vT, int lane, int n0,
                                              const v8f& acc) {
  const int e  = n0 + (lane & 15);
  const int r0 = (lane >> 4) << 3;
  U4 d;
  d.x = pk_bf16(acc[0], acc[1]);
  d.y = pk_bf16(acc[2], acc[3]);
  d.z = pk_bf16(acc[4], acc[5]);
  d.w = pk_bf16(acc[6], acc[7]);
  *reinterpret_cast<U4*>(vT + e * 16 + r0) = d;
}

// ---------------------------------------------------------------------------
// Pre-kernel: transpose weights to bf16 [n][k] panels + copy biases (f32).
// ws layout: bf16 WqT|WkT|WvT|WfcT (4 x 65536 halves) then f32 bias[1024].
// ---------------------------------------------------------------------------
__global__ void __launch_bounds__(256) prep_weights_kernel(
    const float* __restrict__ Wq, const float* __restrict__ Wk,
    const float* __restrict__ Wv, const float* __restrict__ Wfc,
    const float* __restrict__ bq, const float* __restrict__ bk,
    const float* __restrict__ bv, const float* __restrict__ bfc,
    uint16_t* __restrict__ wt, float* __restrict__ bias) {
  const int tid = blockIdx.x * 256 + threadIdx.x;
  if (tid >= 4 * 65536) return;
  const int idx = tid & 65535;
  const int n = idx >> 8;     // output column (head-major h*64+e, or d for fc)
  const int k = idx & 255;    // reduction dim
  if (tid < 3 * 65536) {
    const int region = tid >> 16;
    const float* W = (region == 0) ? Wq : ((region == 1) ? Wk : Wv);
    // W[h][k][e] with h = n>>6, e = n&63
    wt[tid] = f2bf(W[(size_t)(n >> 6) * (DMODEL * DH) + (size_t)k * DH + (n & 63)]);
  } else {
    wt[tid] = f2bf(Wfc[(size_t)k * DMODEL + n]);   // WfcT[d][c]
  }
  if (tid < 256)        bias[tid] = bq[tid];
  else if (tid < 512)   bias[tid] = bk[tid - 256];
  else if (tid < 768)   bias[tid] = bv[tid - 512];
  else if (tid < 1024)  bias[tid] = bfc[tid - 768];
}

// ---------------------------------------------------------------------------
// Main fused kernel. One wave = one 16-row tile = 2 words of 8 tokens.
// Wave-private LDS slab (32 KB), byte offsets within the slab:
//   [    0,  8192)  x_lds  bf16 [16][256]   (reused as out_lds)
//   [ 8192, 24576)  fp32 x staging buffer [16][256]  -- dead after convert,
//                   then aliased by k_lds [8192..16384) / q_lds [16384..24576)
//   [24576, 32768)  vT bf16 [256][16]
// ---------------------------------------------------------------------------
__global__ void __launch_bounds__(WAVES * 32, 1) attn_kernel(
    const float* __restrict__ x, const uint16_t* __restrict__ wt,
    const float* __restrict__ bias, float* __restrict__ out) {
  extern __shared__ char smem_raw[];
  const int lane = threadIdx.x & 31;
  const int wid  = threadIdx.x >> 5;
  const int lo16 = lane & 15;
  const int r0   = (lane >> 4) << 3;       // D-tile row offset for this lane

  uint16_t* wsh   = reinterpret_cast<uint16_t*>(smem_raw) + wid * 16384;
  uint16_t* x_lds = wsh;                   // [16][256] bf16, reused as out_lds
  uint16_t* k_lds = wsh + 4096;            // [16][256]
  uint16_t* q_lds = wsh + 8192;            // [16][256]
  uint16_t* vT    = wsh + 12288;           // [256][16] (per-head [64][16])

  const size_t rbase = ((size_t)blockIdx.x * WAVES + wid) * 16;  // token row

  // Pull the (L2-resident) weight panels toward this WGP.
  __builtin_prefetch(wt + (size_t)lane * 256, 0, 3);

  // ---- Stage x tile (16 rows x 256 f32 = 16 KB) into LDS -------------------
#if USE_ASYNC_STAGE
  {
    // Async copy raw fp32 bytes global->LDS (no VGPR round trip, ASYNCcnt).
    const uint32_t stage_base = (uint32_t)wid * 32768u + 8192u;
    const float* gbase = x + rbase * DMODEL;
    #pragma unroll
    for (int i = 0; i < 32; ++i) {
      const uint32_t off = (uint32_t)lane * 16u + (uint32_t)i * 512u;
      const uint32_t lds_addr = stage_base + off;
      asm volatile("global_load_async_to_lds_b128 %0, %1, %2"
                   :: "v"(lds_addr), "v"(off), "s"(gbase)
                   : "memory");
    }
    asm volatile("s_wait_asynccnt 0x0" ::: "memory");
  }
  // Convert fp32 (LDS staging) -> bf16 x_lds.
  {
    const float* xs = reinterpret_cast<const float*>(smem_raw + wid * 32768 + 8192);
    const int r  = lane >> 1;
    const int ch = (lane & 1) * 128;
    const float* xg = xs + r * DMODEL + ch;
    uint16_t*    xl = x_lds + r * DMODEL + ch;
    #pragma unroll
    for (int c = 0; c < 128; c += 8) {
      F4 f0 = *reinterpret_cast<const F4*>(xg + c);
      F4 f1 = *reinterpret_cast<const F4*>(xg + c + 4);
      U4 d;
      d.x = pk_bf16(f0.x, f0.y); d.y = pk_bf16(f0.z, f0.w);
      d.z = pk_bf16(f1.x, f1.y); d.w = pk_bf16(f1.z, f1.w);
      *reinterpret_cast<U4*>(xl + c) = d;
    }
  }
#else
  {
    const int r  = lane >> 1;
    const int ch = (lane & 1) * 128;
    const float* xg = x + (rbase + r) * DMODEL + ch;
    uint16_t*    xl = x_lds + r * DMODEL + ch;
    #pragma unroll
    for (int c = 0; c < 128; c += 8) {
      F4 f0 = *reinterpret_cast<const F4*>(xg + c);
      F4 f1 = *reinterpret_cast<const F4*>(xg + c + 4);
      U4 d;
      d.x = pk_bf16(f0.x, f0.y); d.y = pk_bf16(f0.z, f0.w);
      d.z = pk_bf16(f1.x, f1.y); d.w = pk_bf16(f1.z, f1.w);
      *reinterpret_cast<U4*>(xl + c) = d;
    }
  }
#endif
  __syncthreads();

  // ---- QKV projections: [16x256] x [256x256] per matrix --------------------
  for (int mtx = 0; mtx < 3; ++mtx) {
    const uint16_t* W = wt + mtx * 65536;
    for (int nt = 0; nt < 16; ++nt) {
      const int n0 = nt * 16;
      v8f acc = vzero();
      #pragma unroll
      for (int ks = 0; ks < 8; ++ks) {
        Frag a = load_a16(x_lds, lane, ks * 32);
        Frag b = load_b16(W, lane, n0, ks * 32);
        acc = __builtin_amdgcn_wmma_f32_16x16x32_bf16(
            false, a.v, false, b.v, (short)0, acc, false, false);
      }
      const float bcol = bias[mtx * 256 + n0 + lo16];
      #pragma unroll
      for (int r = 0; r < 8; ++r) acc[r] += bcol;
      if (mtx == 0)      store_tile_rm(q_lds, lane, n0, acc);
      else if (mtx == 1) store_tile_rm(k_lds, lane, n0, acc);
      else               store_tile_vT(vT, lane, n0, acc);
    }
  }
  __syncthreads();

  // ---- Attention per head: scores^T = K x Q^T, softmax, wei @ V ------------
  const float kscale = 0.125f;             // 1/sqrt(DH)
  for (int h = 0; h < NHEADS; ++h) {
    const uint16_t* kh = k_lds + h * DH;
    const uint16_t* qh = q_lds + h * DH;
    v8f sc = vzero();
    #pragma unroll
    for (int ks = 0; ks < 2; ++ks) {       // K = DH = 64 -> 2 steps
      Frag a = load_a16(kh, lane, ks * 32);          // A[s][e] = k
      Frag b = load_b16(qh, lane, 0, ks * 32);       // B[e][t] = q^T
      sc = __builtin_amdgcn_wmma_f32_16x16x32_bf16(
          false, a.v, false, b.v, (short)0, sc, false, false);
    }
    // sc = scores^T tile: lane column t = lo16, rows s = r0+r.
    const int t = lo16;
    float mx = -3.0e38f;
    #pragma unroll
    for (int r = 0; r < 8; ++r) {
      const int s = r0 + r;
      float val = sc[r] * kscale;
      const bool ok = ((s >> 3) == (t >> 3)) && ((s & 7) <= (t & 7));  // causal
      val = ok ? val : -3.0e38f;
      sc[r] = val;
      mx = fmaxf(mx, val);
    }
    mx = fmaxf(mx, __shfl_xor(mx, 16));    // other half of the s range
    float sum = 0.f;
    #pragma unroll
    for (int r = 0; r < 8; ++r) {
      const float e = __expf(sc[r] - mx);
      sc[r] = e;
      sum += e;
    }
    sum += __shfl_xor(sum, 16);
    const float inv = 1.0f / sum;

    // wei lands directly in A-fragment layout (m=t, K=s; upper K half zero).
    Frag aw;
    #pragma unroll
    for (int j = 0; j < 4; ++j) aw.w[j] = pk_bf16(sc[2 * j] * inv, sc[2 * j + 1] * inv);
    aw.w[4] = aw.w[5] = aw.w[6] = aw.w[7] = 0u;

    const uint16_t* vh = vT + h * DH * 16;
    #pragma unroll
    for (int et = 0; et < 4; ++et) {       // N = DH = 64 -> 4 tiles
      Frag bv = load_bv(vh, lane, et * 16);
      v8f o = __builtin_amdgcn_wmma_f32_16x16x32_bf16(
          false, aw.v, false, bv.v, (short)0, vzero(), false, false);
      store_tile_rm(x_lds /* out_lds */, lane, h * DH + et * 16, o);
    }
  }
  __syncthreads();

  // ---- Output projection: [16x256] x WfcT + bfc ----------------------------
  const uint16_t* Wf = wt + 3 * 65536;
  const float*    bf = bias + 768;
  for (int dt = 0; dt < 16; ++dt) {
    const int d0 = dt * 16;
    v8f acc = vzero();
    #pragma unroll
    for (int ks = 0; ks < 8; ++ks) {
      Frag a = load_a16(x_lds, lane, ks * 32);
      Frag b = load_b16(Wf, lane, d0, ks * 32);
      acc = __builtin_amdgcn_wmma_f32_16x16x32_bf16(
          false, a.v, false, b.v, (short)0, acc, false, false);
    }
    const int col = d0 + lo16;
    const float bb = bf[col];
    float* op = out + (rbase + r0) * DMODEL + col;
    #pragma unroll
    for (int r = 0; r < 8; ++r) op[(size_t)r * DMODEL] = acc[r] + bb;
  }
}

extern "C" void kernel_launch(void* const* d_in, const int* in_sizes, int n_in,
                              void* d_out, int out_size, void* d_ws, size_t ws_size,
                              hipStream_t stream) {
  (void)in_sizes; (void)n_in; (void)out_size; (void)ws_size;
  const float* x   = (const float*)d_in[0];
  const float* Wq  = (const float*)d_in[1];
  const float* bq  = (const float*)d_in[2];
  const float* Wk  = (const float*)d_in[3];
  const float* bk  = (const float*)d_in[4];
  const float* Wv  = (const float*)d_in[5];
  const float* bv  = (const float*)d_in[6];
  const float* Wfc = (const float*)d_in[7];
  const float* bfc = (const float*)d_in[8];
  float* out = (float*)d_out;

  // d_ws: 4*65536 bf16 transposed weight panels (512 KB) + 1024 f32 biases.
  uint16_t* wt   = (uint16_t*)d_ws;
  float*    bias = (float*)((char*)d_ws + (size_t)4 * 65536 * sizeof(uint16_t));

  prep_weights_kernel<<<1024, 256, 0, stream>>>(Wq, Wk, Wv, Wfc, bq, bk, bv, bfc,
                                                wt, bias);

  const int blocks = BATCH / (2 * WAVES);        // 4096 blocks, 2 words/wave
  const size_t shmem = (size_t)WAVES * 32768;    // 32 KB wave-private LDS
  attn_kernel<<<blocks, WAVES * 32, shmem, stream>>>(x, wt, bias, out);
}
